// grad_18038862643482
// MI455X (gfx1250) — matrix operand order repaired and checked by reading, not compiled, not run on hardware
//
#include <hip/hip_runtime.h>
#include <hip/hip_bf16.h>
#include <stdint.h>

// ---------------------------------------------------------------------------
// MLP gradient:  out = -W1 * ((W2 * (W3 .* (1-tanh^2(L2)))) .* (1-tanh^2(L1)))
// Feature-major GEMMs: D[feat,16batch] = A_weights x B_activations.
// Weights are prepacked (d_ws) into per-lane A-fragment bytes in the EXACT
// order they are consumed (frag 0..1279); the kernel streams the 1.25MB pack
// sequentially through a double-buffered 2x16KB LDS stage shared by all 4
// waves of the block (async global->LDS copies overlap the WMMAs).
// A-fragment ds_loads are software-pipelined depth-2 ahead of the WMMAs.
// Activations flow C-tile -> XOR16 lane swizzle -> B-fragment in registers.
// ---------------------------------------------------------------------------

typedef _Float16 v16h  __attribute__((ext_vector_type(16)));
typedef float    v8f   __attribute__((ext_vector_type(8)));
typedef float    f4    __attribute__((ext_vector_type(4)));
typedef uint32_t u32x4 __attribute__((ext_vector_type(4)));
typedef int      v4i   __attribute__((ext_vector_type(4)));

// Pointer element types for the async-copy builtin:
// param0 = int4 addrspace(1) (global), param1 = int4 addrspace(3) (LDS).
typedef __attribute__((address_space(1))) v4i as1_v4i;
typedef __attribute__((address_space(3))) v4i as3_v4i;

union H16 { v16h h; uint32_t u[8]; };
union HP  { _Float16 h[2]; uint32_t u; };

#ifndef __has_builtin
#define __has_builtin(x) 0
#endif

#if __has_builtin(__builtin_amdgcn_global_load_async_to_lds_b128)
#define HAVE_ASYNC_LDS 1
#else
#define HAVE_ASYNC_LDS 0
#endif

#if __has_builtin(__builtin_amdgcn_s_wait_asynccnt)
#define WAIT_ASYNC0() __builtin_amdgcn_s_wait_asynccnt(0)
#else
#define WAIT_ASYNC0() asm volatile("s_wait_asynccnt 0" ::: "memory")
#endif

__device__ __forceinline__ uint32_t pkh(float a, float b) {
  HP p; p.h[0] = (_Float16)a; p.h[1] = (_Float16)b; return p.u;
}

__device__ __forceinline__ float fast_tanh(float x) {
  // tanh(x) = 1 - 2/(exp2(2x*log2e)+1)   (v_exp_f32 + v_rcp_f32)
  float e = __builtin_amdgcn_exp2f(x * 2.885390081777927f);
  return 1.0f - 2.0f * __builtin_amdgcn_rcpf(1.0f + e);
}

// ds_swizzle imm: group-of-32, xor_mask=0x10, and_mask=0x1f  -> lane ^ 16
#define SWZ_SWAPX16 0x401f

// Two adjacent 16x16 C-tiles (f16 pairs packed in pend/cur, 4 dwords each)
// -> one 32x16 B-fragment for the next WMMA. Cross-half data moves via XOR16.
__device__ __forceinline__ void pair_to_bfrag(const uint32_t* pend,
                                              const uint32_t* cur,
                                              bool loHalf, H16& o) {
#pragma unroll
  for (int d = 0; d < 4; ++d) {
    uint32_t s  = loHalf ? cur[d] : pend[d];
    uint32_t xv = (uint32_t)__builtin_amdgcn_ds_swizzle((int)s, SWZ_SWAPX16);
    o.u[d]     = loHalf ? pend[d] : xv;   // K = kbase+0..7
    o.u[4 + d] = loHalf ? xv : cur[d];    // K = kbase+8..15
  }
}

// ---------------------------------------------------------------------------
// Weight prepack: A-fragment layout (16x32 f16 A matrix, wave32):
// lane l: m = mt*16 + (l&15); kb = (l<16?0:16);
// halves h=0..15: k = kt*32 + kb + h; 32B per lane per frag.
// Regions in consumption order: A1T(W1^T) frags 0..127, A2T(W2^T) 128..639,
//                               A2(W2) 640..1151, A1(W1) 1152..1279.
// ---------------------------------------------------------------------------
__global__ __launch_bounds__(256) void pack_weights_kernel(
    const float* __restrict__ W1, const float* __restrict__ W2,
    _Float16* __restrict__ dst) {
  int t = blockIdx.x * blockDim.x + threadIdx.x;
  if (t >= 1280 * 32) return;
  int frag = t >> 5, lane = t & 31;

  const float* src; bool trans; int nkt, f;
  if (frag < 128)       { src = W1; trans = true;  nkt = 4;  f = frag;        }
  else if (frag < 640)  { src = W2; trans = true;  nkt = 16; f = frag - 128;  }
  else if (frag < 1152) { src = W2; trans = false; nkt = 16; f = frag - 640;  }
  else                  { src = W1; trans = false; nkt = 16; f = frag - 1152; }

  int mt = f / nkt, kt = f % nkt;
  int m  = mt * 16 + (lane & 15);
  int kb = (lane < 16) ? 0 : 16;

  H16 o;
#pragma unroll
  for (int h = 0; h < 16; ++h) {
    int k = kt * 32 + kb + h;
    float v = trans ? src[(size_t)k * 512 + m] : src[(size_t)m * 512 + k];
    o.h[h] = (_Float16)v;
  }
  *(v16h*)(dst + (size_t)frag * 512 + lane * 16) = o.h;
}

// ---------------------------------------------------------------------------
// Main fused kernel: 1 wave per 16-sample slab, 4 waves/block.
// LDS: [0,32K) = 2 x 16KB weight stream buffers (shared, double buffered)
//      [32K,96K)  = per-wave s1 stash (f16 pairs, lane-private)
//      [96K,160K) = per-wave g2 B-fragment stash
// Weight stream: 80 chunks x 16KB, chunk c = frags [16c, 16c+16).
// 160KB/block -> two blocks per 320KB WGP; the second block's waves cover
// this block's barrier / async-wait stalls.
// ---------------------------------------------------------------------------
#define NCHUNK 80

__global__ __launch_bounds__(128) void mlp_grad_kernel(
    const float* __restrict__ x,  const float* __restrict__ b1,
    const float* __restrict__ b2, const float* __restrict__ w3,
    const _Float16* __restrict__ packs, float* __restrict__ out, int nSlabs) {
  extern __shared__ char smem[];
  const int  tid  = (int)threadIdx.x;
  const int  lane = tid & 31;
  const int  wv   = tid >> 5;
  const int  slab = (int)blockIdx.x * 4 + wv;           // wave-uniform
  const bool valid = slab < nSlabs;
  const int  slabC = valid ? slab : 0;
  const bool lo   = lane < 16;
  const int  hsel = lo ? 0 : 1;
  const int  l15  = lane & 15;

  char* wb0 = smem;
  char* wb1 = smem + 16384;
  char* s1p = smem + 32768 + wv * 16384;
  char* g2p = smem + 98304 + wv * 16384;
  const char* pk = (const char*)packs;

  u32x4 pf[8];  // fallback prefetch registers (dead code in async path)

  auto issue = [&](int c) {
    const char* src = pk + (size_t)c * 16384 + tid * 16;
#if HAVE_ASYNC_LDS
    char* dst = ((c & 1) ? wb1 : wb0) + tid * 16;
#pragma unroll
    for (int i = 0; i < 8; ++i)
      __builtin_amdgcn_global_load_async_to_lds_b128(
          (as1_v4i*)(src + i * 2048), (as3_v4i*)(dst + i * 2048), 0, 0);
#else
#pragma unroll
    for (int i = 0; i < 8; ++i) pf[i] = *(const u32x4*)(src + i * 2048);
#endif
  };
  auto commit = [&](int c) {
#if !HAVE_ASYNC_LDS
    char* dst = ((c & 1) ? wb1 : wb0) + tid * 16;
#pragma unroll
    for (int i = 0; i < 8; ++i) *(u32x4*)(dst + i * 2048) = pf[i];
#else
    (void)c;
#endif
  };
  auto syncpre = [&]() {
#if HAVE_ASYNC_LDS
    WAIT_ASYNC0();
#endif
    __syncthreads();
  };

  // Prologue: bring in chunk 0.
  issue(0);
  commit(0);

  // ---- x -> B-fragments (B layout: lane holds K=kbase..kbase+15, N=l&15)
  const float* xr = x + ((size_t)(slabC * 16 + l15)) * 128;
  v16h bx[4];
#pragma unroll
  for (int kt = 0; kt < 4; ++kt) {
    const f4* xp = (const f4*)(xr + kt * 32 + hsel * 16);
    f4 a0 = xp[0], a1 = xp[1], a2 = xp[2], a3 = xp[3];
    H16 o;
    o.u[0] = pkh(a0[0], a0[1]); o.u[1] = pkh(a0[2], a0[3]);
    o.u[2] = pkh(a1[0], a1[1]); o.u[3] = pkh(a1[2], a1[3]);
    o.u[4] = pkh(a2[0], a2[1]); o.u[5] = pkh(a2[2], a2[3]);
    o.u[6] = pkh(a3[0], a3[1]); o.u[7] = pkh(a3[2], a3[3]);
    bx[kt] = o.h;
  }

  uint32_t pend[4];

  // ===== Layer 1 (chunks 0..7): h1 = tanh(W1^T x + b1); s1 -> LDS
  v16h h1f[16];
#pragma unroll
  for (int fc = 0; fc < 8; ++fc) {
    const int cc = fc;
    syncpre();
    if (cc + 1 < NCHUNK) issue(cc + 1);
    const v16h* wb = (const v16h*)((cc & 1) ? wb1 : wb0);
    // depth-2 pipelined A-fragment stream across the whole 16-frag chunk
    v16h aA = wb[lane], aB = wb[32 + lane];
#pragma unroll
    for (int s = 0; s < 4; ++s) {
      const int ft = fc * 4 + s;
      // hoist bias loads ahead of the WMMA chain
      const f4* bp = (const f4*)(b1 + ft * 16 + hsel * 8);
      f4 bv0 = bp[0], bv1 = bp[1];
      v8f acc = {};
#pragma unroll
      for (int kt = 0; kt < 4; ++kt) {
        const int j = s * 4 + kt;
        v16h aN = aA;
        if (j + 2 < 16) aN = wb[(j + 2) * 32 + lane];
        acc = __builtin_amdgcn_wmma_f32_16x16x32_f16(
            false, aA, false, bx[kt], (short)0, acc, false, false);
        aA = aB; aB = aN;
      }
      uint32_t cur[4], sp[4];
#pragma unroll
      for (int d = 0; d < 4; ++d) {
        int r0 = 2 * d, r1 = 2 * d + 1;
        float z0 = acc[r0] + ((r0 < 4) ? bv0[r0] : bv1[r0 - 4]);
        float z1 = acc[r1] + ((r1 < 4) ? bv0[r1] : bv1[r1 - 4]);
        float t0 = fast_tanh(z0), t1 = fast_tanh(z1);
        cur[d] = pkh(t0, t1);
        sp[d]  = pkh(1.0f - t0 * t0, 1.0f - t1 * t1);
      }
      u32x4 s4 = {sp[0], sp[1], sp[2], sp[3]};
      *(u32x4*)(s1p + ft * 512 + lane * 16) = s4;
      if ((s & 1) == 0) {
        pend[0] = cur[0]; pend[1] = cur[1]; pend[2] = cur[2]; pend[3] = cur[3];
      } else {
        H16 o; pair_to_bfrag(pend, cur, lo, o);
        h1f[ft >> 1] = o.h;
      }
    }
    if (cc + 1 < NCHUNK) commit(cc + 1);
  }

  // ===== Layer 2 (chunks 8..39): g2 = W3 .* (1-tanh^2(W2^T h1 + b2)) -> LDS
#pragma unroll
  for (int ft = 0; ft < 32; ++ft) {
    const int cc = 8 + ft;
    syncpre();
    if (cc + 1 < NCHUNK) issue(cc + 1);
    const v16h* wb = (const v16h*)((cc & 1) ? wb1 : wb0);
    // hoist bias / w3 loads ahead of the WMMA chain
    const f4* bp = (const f4*)(b2 + ft * 16 + hsel * 8);
    const f4* wp = (const f4*)(w3 + ft * 16 + hsel * 8);
    f4 bv0 = bp[0], bv1 = bp[1], wv0 = wp[0], wv1 = wp[1];
    v16h aA = wb[lane], aB = wb[32 + lane];
    v8f acc = {};
#pragma unroll
    for (int kt = 0; kt < 16; ++kt) {
      v16h aN = aA;
      if (kt + 2 < 16) aN = wb[(kt + 2) * 32 + lane];
      acc = __builtin_amdgcn_wmma_f32_16x16x32_f16(
          false, aA, false, h1f[kt], (short)0, acc, false, false);
      aA = aB; aB = aN;
    }
    uint32_t cur[4];
#pragma unroll
    for (int d = 0; d < 4; ++d) {
      int r0 = 2 * d, r1 = 2 * d + 1;
      float z0 = acc[r0] + ((r0 < 4) ? bv0[r0] : bv1[r0 - 4]);
      float z1 = acc[r1] + ((r1 < 4) ? bv0[r1] : bv1[r1 - 4]);
      float t0 = fast_tanh(z0), t1 = fast_tanh(z1);
      float g0 = ((r0 < 4) ? wv0[r0] : wv1[r0 - 4]) * (1.0f - t0 * t0);
      float g1 = ((r1 < 4) ? wv0[r1] : wv1[r1 - 4]) * (1.0f - t1 * t1);
      cur[d] = pkh(g0, g1);
    }
    if ((ft & 1) == 0) {
      pend[0] = cur[0]; pend[1] = cur[1]; pend[2] = cur[2]; pend[3] = cur[3];
    } else {
      H16 o; pair_to_bfrag(pend, cur, lo, o);
      *(v16h*)(g2p + (ft >> 1) * 1024 + lane * 32) = o.h;
    }
    if (cc + 1 < NCHUNK) commit(cc + 1);
  }

  // ===== Backward 1 (chunks 40..71): g1 = (W2 g2) .* s1
  v16h g1f[16];
#pragma unroll
  for (int ft = 0; ft < 32; ++ft) {
    const int cc = 40 + ft;
    syncpre();
    if (cc + 1 < NCHUNK) issue(cc + 1);
    const v16h* wb = (const v16h*)((cc & 1) ? wb1 : wb0);
    // hoist the s1 load ahead of the WMMA chain
    u32x4 sv = *(const u32x4*)(s1p + ft * 512 + lane * 16);
    // weights depth-2, g2 operand depth-1 pipelined
    v16h aA = wb[lane], aB = wb[32 + lane];
    v16h gA = *(const v16h*)(g2p + lane * 32);
    v8f acc = {};
#pragma unroll
    for (int kt = 0; kt < 16; ++kt) {
      v16h aN = aA, gN = gA;
      if (kt + 2 < 16) aN = wb[(kt + 2) * 32 + lane];
      if (kt + 1 < 16) gN = *(const v16h*)(g2p + (kt + 1) * 1024 + lane * 32);
      acc = __builtin_amdgcn_wmma_f32_16x16x32_f16(
          false, aA, false, gA, (short)0, acc, false, false);
      aA = aB; aB = aN; gA = gN;
    }
    uint32_t cur[4];
#pragma unroll
    for (int d = 0; d < 4; ++d) {
      HP q; q.u = sv[d];
      float g0 = acc[2 * d]     * (float)q.h[0];
      float g1 = acc[2 * d + 1] * (float)q.h[1];
      cur[d] = pkh(g0, g1);
    }
    if ((ft & 1) == 0) {
      pend[0] = cur[0]; pend[1] = cur[1]; pend[2] = cur[2]; pend[3] = cur[3];
    } else {
      H16 o; pair_to_bfrag(pend, cur, lo, o);
      g1f[ft >> 1] = o.h;
    }
    if (cc + 1 < NCHUNK) commit(cc + 1);
  }

  // ===== Backward 2 (chunks 72..79): out = -(W1 g1)
#pragma unroll
  for (int dt = 0; dt < 8; ++dt) {
    const int cc = 72 + dt;
    syncpre();
    if (cc + 1 < NCHUNK) issue(cc + 1);
    const v16h* wb = (const v16h*)((cc & 1) ? wb1 : wb0);
    v16h aA = wb[lane], aB = wb[32 + lane];
    v8f acc = {};
#pragma unroll
    for (int kt = 0; kt < 16; ++kt) {
      v16h aN = aA;
      if (kt + 2 < 16) aN = wb[(kt + 2) * 32 + lane];
      acc = __builtin_amdgcn_wmma_f32_16x16x32_f16(
          false, aA, false, g1f[kt], (short)0, acc, false, false);
      aA = aB; aB = aN;
    }
    if (valid) {
      float* op = out + ((size_t)(slab * 16 + l15)) * 128 + dt * 16 + hsel * 8;
      f4 o0 = {-acc[0], -acc[1], -acc[2], -acc[3]};
      f4 o1 = {-acc[4], -acc[5], -acc[6], -acc[7]};
      *(f4*)op       = o0;
      *(f4*)(op + 4) = o1;
    }
    if (cc + 1 < NCHUNK) commit(cc + 1);
  }
}

extern "C" void kernel_launch(void* const* d_in, const int* in_sizes, int n_in,
                              void* d_out, int out_size, void* d_ws, size_t ws_size,
                              hipStream_t stream) {
  (void)n_in; (void)out_size; (void)ws_size;
  const float* x  = (const float*)d_in[0];
  const float* W1 = (const float*)d_in[1];
  const float* b1 = (const float*)d_in[2];
  const float* W2 = (const float*)d_in[3];
  const float* b2 = (const float*)d_in[4];
  const float* W3 = (const float*)d_in[5];
  // b3 (d_in[6]) does not affect the gradient.
  float*     out   = (float*)d_out;
  _Float16*  packs = (_Float16*)d_ws;   // 1280 frags * 1KB = 1.25 MB

  // Prepack weights every call (deterministic; ws is not preserved).
  pack_weights_kernel<<<160, 256, 0, stream>>>(W1, W2, packs);

  int B      = in_sizes[0] / 128;
  int nSlabs = B / 16;                  // 4096
  int blocks = (nSlabs + 3) / 4;        // 4 waves (slabs) per block
  mlp_grad_kernel<<<blocks, 128, 163840, stream>>>(x, b1, b2, W3, packs, out, nSlabs);
}
